// Attention_49349174231571
// MI455X (gfx1250) — matrix operand order repaired and checked by reading, not compiled, not run on hardware
//
#include <hip/hip_runtime.h>
#include <hip/hip_bf16.h>
#include <stdint.h>

typedef __attribute__((ext_vector_type(16))) _Float16 v16h;
typedef __attribute__((ext_vector_type(8)))  _Float16 v8h;
typedef __attribute__((ext_vector_type(8)))  float    v8f;

namespace {
constexpr int B_  = 2;
constexpr int T_  = 2048;
constexpr int D_  = 1024;
constexpr int H_  = 16;
constexpr int HD_ = 64;
constexpr int N_  = B_ * T_;   // 4096 rows
}

// ---------------------------------------------------------------------------
// WMMA fragment loaders (wave32, V_WMMA_F32_16X16X32_F16; contiguous k = 16g+j order, not the ISA table order)
// ---------------------------------------------------------------------------

// A matrix 16x32 f16 (MxK), row-major source with leading dim `ld`.
// Lane (g=lane/16, m=lane%16): slot j -> K = 16g+j (contiguous, matches wmma_bT_frag)
__device__ __forceinline__ v16h wmma_a_frag(const _Float16* __restrict__ base,
                                            int ld, int row0, int k0) {
  const int lane = threadIdx.x & 31;
  const int g = lane >> 4, m = lane & 15;
  const _Float16* p = base + (size_t)(row0 + m) * ld + k0 + 16 * g;
  v8h lo = *(const v8h*)(p);
  v8h hi = *(const v8h*)(p + 8);
  v16h a;
#pragma unroll
  for (int i = 0; i < 8; ++i) { a[i] = lo[i]; a[8 + i] = hi[i]; }
  return a;
}

// B matrix 32x16 f16 (KxN) where memory holds B[k][n] = src[(n0+n)*ld + k0 + k]
// (i.e. source is row-major along K: the W^T / K^T / V^T access pattern).
// Lane (g,n): slot j holds B[16g+j][n] -> one contiguous 32-byte load.
__device__ __forceinline__ v16h wmma_bT_frag(const _Float16* __restrict__ base,
                                             int ld, int n0, int k0) {
  const int lane = threadIdx.x & 31;
  const int g = lane >> 4, n = lane & 15;
  const _Float16* p = base + (size_t)(n0 + n) * ld + k0 + 16 * g;
  return *(const v16h*)p;
}

__device__ __forceinline__ v8f wmma_f32_16x16x32(v16h a, v16h b, v8f c) {
  return __builtin_amdgcn_wmma_f32_16x16x32_f16(false, a, false, b,
                                                (short)0, c, false, false);
}

// 16-lane-half reductions (wave32: masks < 16 stay inside each half)
__device__ __forceinline__ float rowmax16(float v) {
#pragma unroll
  for (int m = 8; m >= 1; m >>= 1) v = fmaxf(v, __shfl_xor(v, m, 32));
  return v;
}
__device__ __forceinline__ float rowsum16(float v) {
#pragma unroll
  for (int m = 8; m >= 1; m >>= 1) v += __shfl_xor(v, m, 32);
  return v;
}

// Async Global -> LDS b128 (ASYNCcnt path, CDNA5 ISA).
// lds_off: wave-relative LDS byte offset; gaddr: 64-bit global address.
__device__ __forceinline__ void async_load_b128(unsigned int lds_off,
                                                unsigned long long gaddr) {
  asm volatile("global_load_async_to_lds_b128 %0, %1, off"
               :: "v"(lds_off), "v"(gaddr) : "memory");
}
__device__ __forceinline__ unsigned int lds_off(const void* p) {
  return (unsigned int)(uintptr_t)p;   // generic LDS addr[31:0] = LDS offset
}

// ---------------------------------------------------------------------------
// Stage 0: f32 -> f16 conversion
// ---------------------------------------------------------------------------
__global__ void cvt_f32_to_f16(const float* __restrict__ src,
                               _Float16* __restrict__ dst, int n) {
  int i = blockIdx.x * blockDim.x + threadIdx.x;
  if (i < n) dst[i] = (_Float16)src[i];
}

// ---------------------------------------------------------------------------
// Stage 1: QKV projection GEMM.  out[row, o] = sum_k X[row,k] * W[o,k] + b[o]
// grid (N/128, D/16, 3), block 256 (8 waves). Wave w owns one 16x16 tile.
// which: 0 -> q into qh [B,H,T,HD]; 1 -> k into kh [B,H,T,HD]; 2 -> v into vt [B,H,HD,T]
// ---------------------------------------------------------------------------
__global__ __launch_bounds__(256)
void qkv_proj_kernel(const _Float16* __restrict__ xq16,
                     const _Float16* __restrict__ xk16,
                     const _Float16* __restrict__ wq16,
                     const _Float16* __restrict__ wk16,
                     const _Float16* __restrict__ wv16,
                     const float* __restrict__ bq,
                     const float* __restrict__ bk,
                     const float* __restrict__ bv,
                     _Float16* __restrict__ qh,
                     _Float16* __restrict__ kh,
                     _Float16* __restrict__ vt) {
  const int w    = threadIdx.x >> 5;
  const int lane = threadIdx.x & 31;
  const int row0 = (blockIdx.x * 8 + w) * 16;
  const int o0   = blockIdx.y * 16;
  const int which = blockIdx.z;

  const _Float16* x   = (which == 0) ? xq16 : xk16;  // k,v project key_states
  const _Float16* wgt = (which == 0) ? wq16 : (which == 1 ? wk16 : wv16);
  const float*    bia = (which == 0) ? bq   : (which == 1 ? bk   : bv);

  v8f acc = {};
#pragma unroll 4
  for (int k0 = 0; k0 < D_; k0 += 32) {
    v16h a = wmma_a_frag(x, D_, row0, k0);
    v16h b = wmma_bT_frag(wgt, D_, o0, k0);   // B[k][n] = W[o0+n][k]
    acc = wmma_f32_16x16x32(a, b, acc);
  }

  const int g = lane >> 4, n = lane & 15;
  const int col = o0 + n;
  const float bias = bia[col];
  const int h  = col / HD_;
  const int hd = col % HD_;
#pragma unroll
  for (int r = 0; r < 8; ++r) {
    const int rowg = row0 + 8 * g + r;         // C/D: lane(g,n), VGPR r -> row 8g+r
    const int bi = rowg / T_, t = rowg % T_;
    const _Float16 val = (_Float16)(acc[r] + bias);
    if (which == 2) {
      vt[(((size_t)bi * H_ + h) * HD_ + hd) * T_ + t] = val;
    } else {
      _Float16* dst = (which == 0) ? qh : kh;
      dst[(((size_t)bi * H_ + h) * T_ + t) * HD_ + hd] = val;
    }
  }
}

// ---------------------------------------------------------------------------
// Stage 2: flash attention. Block = 8 waves covering 8 q-tiles of one (b,h).
// K/V tiles are double-buffered in LDS, streamed with async global->LDS b128
// (ASYNCcnt) and shared by all waves of the block. grid (B*H*16), block 256.
// ---------------------------------------------------------------------------
__global__ __launch_bounds__(256)
void flash_attn_kernel(const _Float16* __restrict__ qh,
                       const _Float16* __restrict__ kh,
                       const _Float16* __restrict__ vt,
                       const float* __restrict__ query_mask,
                       const float* __restrict__ key_mask,
                       const int* __restrict__ mask_future_p,
                       float* __restrict__ out) {
  __shared__ __align__(32) _Float16 Kt[2][32][64];    // [buf][key][d]   2 x 4 KB
  __shared__ __align__(32) _Float16 Vt2[2][64][32];   // [buf][d][key]   2 x 4 KB
  __shared__ __align__(32) _Float16 Ptile[8][16][32]; // per-wave P      8 KB

  const int tid  = threadIdx.x;
  const int w    = tid >> 5;
  const int lane = tid & 31;
  const int bh   = blockIdx.x >> 4;        // 16 blocks per (b,h)
  const int qblk = blockIdx.x & 15;
  const int b    = bh >> 4;                // / H
  const int h    = bh & 15;
  const int q0   = (qblk * 8 + w) * 16;    // this wave's q-tile
  const int mf   = mask_future_p[0];

  const _Float16* qb = qh + (size_t)bh * T_ * HD_;  // [T, 64]
  const _Float16* kb = kh + (size_t)bh * T_ * HD_;  // [T, 64]
  const _Float16* vb = vt + (size_t)bh * HD_ * T_;  // [64, T] (transposed)

  const int g = lane >> 4, n = lane & 15;

  // Per-thread async staging slots: 256 threads x 16B == one 4 KB tile each.
  const int krow = tid >> 3, kchunk = tid & 7;   // K tile: 32 rows x 128 B
  const int vrow = tid >> 2, vchunk = tid & 3;   // V tile: 64 rows x  64 B

  // Q fragments for both 32-wide d chunks (reused across all key steps)
  const v16h aQ0 = wmma_a_frag(qb, HD_, q0, 0);
  const v16h aQ1 = wmma_a_frag(qb, HD_, q0, 32);

  float mrow[8], lrow[8];
  v8f oc0 = {}, oc1 = {}, oc2 = {}, oc3 = {};   // 16x64 f32 output accumulator
#pragma unroll
  for (int r = 0; r < 8; ++r) { mrow[r] = -1e30f; lrow[r] = 0.0f; }

  const int kend_blk = mf ? (qblk * 8 + 8) * 16 : T_;  // causal bound of last wave
  const int kend_wav = mf ? (q0 + 16) : T_;            // this wave's causal bound
  const int nsteps   = kend_blk / 32;

  // Issue tile 0 (2 async instructions per wave -> ASYNCcnt == 2)
  async_load_b128(lds_off(&Kt[0][krow][kchunk * 8]),
                  (unsigned long long)(uintptr_t)(kb + (size_t)krow * HD_ + kchunk * 8));
  async_load_b128(lds_off(&Vt2[0][vrow][vchunk * 8]),
                  (unsigned long long)(uintptr_t)(vb + (size_t)vrow * T_ + vchunk * 8));

  for (int s = 0; s < nsteps; ++s) {
    const int key0 = s * 32;
    const int buf  = s & 1;
    if (s + 1 < nsteps) {
      // Prefetch next K/V tile into the other buffer, then retire current tile.
      const int nk = key0 + 32, nb = buf ^ 1;
      async_load_b128(lds_off(&Kt[nb][krow][kchunk * 8]),
                      (unsigned long long)(uintptr_t)(kb + (size_t)(nk + krow) * HD_ + kchunk * 8));
      async_load_b128(lds_off(&Vt2[nb][vrow][vchunk * 8]),
                      (unsigned long long)(uintptr_t)(vb + (size_t)vrow * T_ + nk + vchunk * 8));
      asm volatile("s_wait_asynccnt 0x2" ::: "memory");
    } else {
      asm volatile("s_wait_asynccnt 0x0" ::: "memory");
    }
    __syncthreads();   // current tile visible to all 8 waves

    if (key0 < kend_wav) {
      const _Float16* kl = &Kt[buf][0][0];   // [32][64] f16 in LDS
      const _Float16* vl = &Vt2[buf][0][0];  // [64][32] f16 in LDS

      // ---- S = Q K^T for 16 q-rows x 32 keys (4 WMMAs, B operands from LDS) ----
      v8f s0 = {}, s1 = {};
      {
        v16h b0 = wmma_bT_frag(kl, HD_, 0, 0);
        s0 = wmma_f32_16x16x32(aQ0, b0, s0);
        v16h b1 = wmma_bT_frag(kl, HD_, 0, 32);
        s0 = wmma_f32_16x16x32(aQ1, b1, s0);
        v16h b2 = wmma_bT_frag(kl, HD_, 16, 0);
        s1 = wmma_f32_16x16x32(aQ0, b2, s1);
        v16h b3 = wmma_bT_frag(kl, HD_, 16, 32);
        s1 = wmma_f32_16x16x32(aQ1, b3, s1);
      }

      const int keyA = key0 + n;
      const int keyB = key0 + 16 + n;
      const float kmA = (1.0f - key_mask[b * T_ + keyA]) * -10000.0f;
      const float kmB = (1.0f - key_mask[b * T_ + keyB]) * -10000.0f;

      // ---- online softmax update, rows r (global row = q0 + 8g + r) ----
#pragma unroll
      for (int r = 0; r < 8; ++r) {
        const int qrow = q0 + 8 * g + r;
        float sa = s0[r] * 0.125f + kmA + ((mf && keyA > qrow) ? -10000.0f : 0.0f);
        float sb = s1[r] * 0.125f + kmB + ((mf && keyB > qrow) ? -10000.0f : 0.0f);
        const float tmax = rowmax16(fmaxf(sa, sb));
        const float newm = fmaxf(mrow[r], tmax);
        const float alpha = __expf(mrow[r] - newm);
        const float pa = __expf(sa - newm);
        const float pb = __expf(sb - newm);
        lrow[r] = lrow[r] * alpha + rowsum16(pa + pb);
        mrow[r] = newm;
        oc0[r] *= alpha; oc1[r] *= alpha; oc2[r] *= alpha; oc3[r] *= alpha;
        // transpose S (row-across-lanes) -> P (row-per-lane A operand) via LDS
        Ptile[w][8 * g + r][n]      = (_Float16)pa;
        Ptile[w][8 * g + r][16 + n] = (_Float16)pb;
      }

      // per-wave LDS is in-order; make the store->load hazard explicit
      asm volatile("s_wait_dscnt 0x0" ::: "memory");

      // ---- O += P V : A = P (16x32), B[key][d] = Vt[d][key] (4 WMMAs) ----
      const _Float16* pbase = &Ptile[w][0][0];
      v16h aP = wmma_a_frag(pbase, 32, 0, 0);
      v16h bV0 = wmma_bT_frag(vl, 32, 0, 0);
      oc0 = wmma_f32_16x16x32(aP, bV0, oc0);
      v16h bV1 = wmma_bT_frag(vl, 32, 16, 0);
      oc1 = wmma_f32_16x16x32(aP, bV1, oc1);
      v16h bV2 = wmma_bT_frag(vl, 32, 32, 0);
      oc2 = wmma_f32_16x16x32(aP, bV2, oc2);
      v16h bV3 = wmma_bT_frag(vl, 32, 48, 0);
      oc3 = wmma_f32_16x16x32(aP, bV3, oc3);
    }

    __syncthreads();   // everyone done reading buf before it is overwritten
  }

  // ---- normalize, apply post-softmax query mask, store [B,T,D] f32 ----
#pragma unroll
  for (int r = 0; r < 8; ++r) {
    const int qrow = q0 + 8 * g + r;
    const float inv = (lrow[r] > 0.0f) ? (1.0f / lrow[r]) : 0.0f;
    const float scale = inv * query_mask[b * T_ + qrow];
    const size_t base = ((size_t)b * T_ + qrow) * D_ + h * HD_;
    out[base + 0  + n] = oc0[r] * scale;
    out[base + 16 + n] = oc1[r] * scale;
    out[base + 32 + n] = oc2[r] * scale;
    out[base + 48 + n] = oc3[r] * scale;
  }
}

// ---------------------------------------------------------------------------
// Host launcher
// ---------------------------------------------------------------------------
extern "C" void kernel_launch(void* const* d_in, const int* in_sizes, int n_in,
                              void* d_out, int out_size, void* d_ws, size_t ws_size,
                              hipStream_t stream) {
  const float* xq  = (const float*)d_in[0];   // query_states [B,T,D]
  const float* xk  = (const float*)d_in[1];   // key_states   [B,T,D]
  const float* qm  = (const float*)d_in[2];   // query_mask   [B,T]
  const float* km  = (const float*)d_in[3];   // key_mask     [B,T]
  const float* Wq  = (const float*)d_in[4];
  const float* bq  = (const float*)d_in[5];
  const float* Wk  = (const float*)d_in[6];
  const float* bk  = (const float*)d_in[7];
  const float* Wv  = (const float*)d_in[8];
  const float* bv  = (const float*)d_in[9];
  const int*   mfp = (const int*)d_in[10];
  float* out = (float*)d_out;

  // Workspace carve-up (f16 elements). Total = 46 MiB.
  _Float16* xq16 = (_Float16*)d_ws;
  _Float16* xk16 = xq16 + (size_t)N_ * D_;          //  8 MiB each
  _Float16* wq16 = xk16 + (size_t)N_ * D_;
  _Float16* wk16 = wq16 + (size_t)D_ * D_;          //  2 MiB each
  _Float16* wv16 = wk16 + (size_t)D_ * D_;
  _Float16* qh   = wv16 + (size_t)D_ * D_;          //  8 MiB each
  _Float16* kh   = qh   + (size_t)B_ * H_ * T_ * HD_;
  _Float16* vt   = kh   + (size_t)B_ * H_ * T_ * HD_;
  const size_t need = ((size_t)vt + (size_t)B_ * H_ * T_ * HD_ * sizeof(_Float16))
                      - (size_t)d_ws;
  if (ws_size < need) return;  // deterministic no-op if scratch too small

  // Stage 0: conversions
  const int nX = N_ * D_;
  const int nW = D_ * D_;
  cvt_f32_to_f16<<<(nX + 255) / 256, 256, 0, stream>>>(xq, xq16, nX);
  cvt_f32_to_f16<<<(nX + 255) / 256, 256, 0, stream>>>(xk, xk16, nX);
  cvt_f32_to_f16<<<(nW + 255) / 256, 256, 0, stream>>>(Wq, wq16, nW);
  cvt_f32_to_f16<<<(nW + 255) / 256, 256, 0, stream>>>(Wk, wk16, nW);
  cvt_f32_to_f16<<<(nW + 255) / 256, 256, 0, stream>>>(Wv, wv16, nW);

  // Stage 1: QKV projections (8 waves/block, 16x16 tile per wave)
  dim3 gproj(N_ / (16 * 8), D_ / 16, 3);
  qkv_proj_kernel<<<gproj, 256, 0, stream>>>(xq16, xk16, wq16, wk16, wv16,
                                             bq, bk, bv, qh, kh, vt);

  // Stage 2: flash attention (block = 8 q-tiles of one (b,h), async K/V staging)
  flash_attn_kernel<<<B_ * H_ * 16, 256, 0, stream>>>(qh, kh, vt, qm, km, mfp, out);
}
